// GraphDiffNetBipartite_24000277250262
// MI455X (gfx1250) — compile-verified, matrix operand order checked by reading, not run on hardware
//
#include <hip/hip_runtime.h>

// ---------------------------------------------------------------------------
// GraphDiffNetBipartite forward for MI455X (gfx1250), bf16 WMMA pipeline.
//
// Roofline: dominant GEMMs are gradX/gradY @ xd (2 x [8192^2]x[8192x64] per
// block) and Wb/Wb^T bipartite matmuls.  ~105 GFLOP total; unique streamed
// A-operands in bf16 = ~1.7 GB per call -> ~73 us floor at 23.3 TB/s HBM.
// All B operands are <= 64x8192 bf16 (<=1 MB) -> resident in the 192 MB L2.
// Matmuls run on v_wmma_f32_16x16x32_bf16 (fp32 accum), 16x64 tile per wave,
// two-stage ping-pong software pipeline (no register-copy rotation), split-K
// (global_atomic_add_f32 reduction) to spread the A-stream over enough WGPs
// to saturate HBM.
// ---------------------------------------------------------------------------

typedef __bf16 bf16_t;
typedef bf16_t v16bf __attribute__((ext_vector_type(16)));
typedef bf16_t v8bf  __attribute__((ext_vector_type(8)));
typedef bf16_t v4bf  __attribute__((ext_vector_type(4)));
typedef float  v8f   __attribute__((ext_vector_type(8)));

#define NV 8192
#define NG 4096
#define KEIG 128
#define CIN 128
#define DW 64
#define NBLK 4
#define NE 65536

__device__ __forceinline__ v16bf ldfrag(const bf16_t* __restrict__ row, int k, int half)
{
    v8bf lo = *(const v8bf*)(row + k + 8 * half);
    v8bf hi = *(const v8bf*)(row + k + 16 + 8 * half);
    return __builtin_shufflevector(lo, hi, 0,1,2,3,4,5,6,7,8,9,10,11,12,13,14,15);
}

#define WMMA_BF16(A_, B_, C_) \
    __builtin_amdgcn_wmma_f32_16x16x32_bf16(false, (A_), false, (B_), (short)0, (C_), false, false)

// ---------------------------------------------------------------------------
// WMMA GEMM, output N=64, C fp32 [M,64].
//   atomicOut=0: C = act( alpha*(A@B) + Cin? + bias? )          (grid.y == 1)
//   atomicOut=1: C += alpha*(A@B) via global_atomic_add_f32     (split-K,
//                grid.y = K/kChunk, C must be pre-zeroed; no Cin/bias/relu)
// A: bf16 row-major [M,K].  BT: bf16 [64,K] (B column-major).
// Wave tile 16(M) x 64(N): 1 A fragment reused across 4 WMMAs.
// kChunk must be a multiple of 64 (two-stage ping-pong pipeline).
// CDNA5 16-bit operand layout: lane l holds row (A) / column (B) l&15; its 16
// elements are the contiguous runs k+8*(l>>4)..+7 and k+16+8*(l>>4)..+7.
// ---------------------------------------------------------------------------
__global__ __launch_bounds__(256)
void k_gemm_n64(const bf16_t* __restrict__ A, const bf16_t* __restrict__ BT,
                float* __restrict__ C, const float* __restrict__ Cin,
                const float* __restrict__ bias, int M, int K, int kChunk,
                float alpha, int relu, int atomicOut)
{
    const int wave = (int)((blockIdx.x * blockDim.x + threadIdx.x) >> 5);
    const int lane = (int)(threadIdx.x & 31);
    const int m0 = wave * 16;
    if (m0 >= M) return;                       // wave-uniform: EXEC stays full
    const int half = lane >> 4;
    const int l16  = lane & 15;
    const int kbeg = (int)blockIdx.y * kChunk;
    const int kend = kbeg + kChunk;            // kChunk divides K, 64 | kChunk

    const bf16_t* __restrict__ arow  = A  + (size_t)(m0 + l16) * (size_t)K;
    const bf16_t* __restrict__ brow0 = BT + (size_t)(l16 +  0) * (size_t)K;
    const bf16_t* __restrict__ brow1 = BT + (size_t)(l16 + 16) * (size_t)K;
    const bf16_t* __restrict__ brow2 = BT + (size_t)(l16 + 32) * (size_t)K;
    const bf16_t* __restrict__ brow3 = BT + (size_t)(l16 + 48) * (size_t)K;

    v8f acc0 = {}, acc1 = {}, acc2 = {}, acc3 = {};

    // Two-stage ping-pong pipeline: set S0 covers k, set S1 covers k+32.
    // Each fragment register is written by exactly one load and read by
    // exactly one WMMA per iteration -> no rotation copies, loads stay two
    // K-steps ahead of their consuming WMMA.
    v16bf a0  = ldfrag(arow,  kbeg, half);
    v16bf b00 = ldfrag(brow0, kbeg, half);
    v16bf b01 = ldfrag(brow1, kbeg, half);
    v16bf b02 = ldfrag(brow2, kbeg, half);
    v16bf b03 = ldfrag(brow3, kbeg, half);
    v16bf a1  = ldfrag(arow,  kbeg + 32, half);
    v16bf b10 = ldfrag(brow0, kbeg + 32, half);
    v16bf b11 = ldfrag(brow1, kbeg + 32, half);
    v16bf b12 = ldfrag(brow2, kbeg + 32, half);
    v16bf b13 = ldfrag(brow3, kbeg + 32, half);

    int k = kbeg;
    for (; k + 64 < kend; k += 64) {
        if (((k & 255) == 0) && (k + 576 <= kend))
            __builtin_prefetch((const void*)(arow + k + 576), 0, 1);
        // consume S0 (k), then refill S0 for k+64
        acc0 = WMMA_BF16(a0, b00, acc0);
        acc1 = WMMA_BF16(a0, b01, acc1);
        acc2 = WMMA_BF16(a0, b02, acc2);
        acc3 = WMMA_BF16(a0, b03, acc3);
        a0  = ldfrag(arow,  k + 64, half);
        b00 = ldfrag(brow0, k + 64, half);
        b01 = ldfrag(brow1, k + 64, half);
        b02 = ldfrag(brow2, k + 64, half);
        b03 = ldfrag(brow3, k + 64, half);
        // consume S1 (k+32), then refill S1 for k+96
        acc0 = WMMA_BF16(a1, b10, acc0);
        acc1 = WMMA_BF16(a1, b11, acc1);
        acc2 = WMMA_BF16(a1, b12, acc2);
        acc3 = WMMA_BF16(a1, b13, acc3);
        a1  = ldfrag(arow,  k + 96, half);
        b10 = ldfrag(brow0, k + 96, half);
        b11 = ldfrag(brow1, k + 96, half);
        b12 = ldfrag(brow2, k + 96, half);
        b13 = ldfrag(brow3, k + 96, half);
    }
    // final two K-steps: fragments already resident
    acc0 = WMMA_BF16(a0, b00, acc0);
    acc1 = WMMA_BF16(a0, b01, acc1);
    acc2 = WMMA_BF16(a0, b02, acc2);
    acc3 = WMMA_BF16(a0, b03, acc3);
    acc0 = WMMA_BF16(a1, b10, acc0);
    acc1 = WMMA_BF16(a1, b11, acc1);
    acc2 = WMMA_BF16(a1, b12, acc2);
    acc3 = WMMA_BF16(a1, b13, acc3);

    // C/D layout: VGPR r -> row m0 + r + 8*half, column = nt*16 + l16.
    v8f accs[4] = {acc0, acc1, acc2, acc3};
    if (atomicOut) {
#pragma unroll
        for (int nt = 0; nt < 4; ++nt)
#pragma unroll
            for (int r = 0; r < 8; ++r) {
                const int row = m0 + r + 8 * half;
                const int col = nt * 16 + l16;
                unsafeAtomicAdd(&C[(size_t)row * 64 + col], alpha * accs[nt][r]);
            }
    } else {
#pragma unroll
        for (int nt = 0; nt < 4; ++nt)
#pragma unroll
            for (int r = 0; r < 8; ++r) {
                const int row = m0 + r + 8 * half;
                const int col = nt * 16 + l16;
                float v = alpha * accs[nt][r];
                if (Cin)  v += Cin[(size_t)row * 64 + col];
                if (bias) v += bias[col];
                if (relu) v = fmaxf(v, 0.0f);
                C[(size_t)row * 64 + col] = v;
            }
    }
}

// --------------------------- pre/elementwise kernels ------------------------

// Fused cdist + cutoff + exp, writes bf16 Wb [NV,NG] and Wb^T [NG,NV].
__global__ __launch_bounds__(256)
void k_wb(const float* __restrict__ vtx, const float* __restrict__ gpos,
          bf16_t* __restrict__ Wb, bf16_t* __restrict__ WbT)
{
    __shared__ float tile[32][33];
    const int tx = threadIdx.x, ty = threadIdx.y;      // 32 x 8
    const int v0 = blockIdx.y * 32, g0 = blockIdx.x * 32;
#pragma unroll
    for (int i = 0; i < 32; i += 8) {
        const int v = v0 + ty + i, g = g0 + tx;
        const float dx = vtx[v*3+0] - gpos[g*3+0];
        const float dy = vtx[v*3+1] - gpos[g*3+1];
        const float dz = vtx[v*3+2] - gpos[g*3+2];
        const float d  = sqrtf(fmaxf(dx*dx + dy*dy + dz*dz, 0.0f));
        const float w  = (d < 8.0f) ? __expf(-d * 0.25f) : 0.0f;
        tile[ty + i][tx] = w;
        Wb[(size_t)v * NG + g] = (bf16_t)w;
    }
    __syncthreads();
#pragma unroll
    for (int i = 0; i < 32; i += 8) {
        const int g = g0 + ty + i, v = v0 + tx;
        WbT[(size_t)g * NV + v] = (bf16_t)tile[tx][ty + i];
    }
}

// deg[row] = 1 + sum(bf16 row); rinv = rsqrt(deg)
__global__ __launch_bounds__(256)
void k_rowsum(const bf16_t* __restrict__ W, int cols,
              float* __restrict__ deg, float* __restrict__ rinv)
{
    __shared__ float red[256];
    const int row = blockIdx.x, tid = threadIdx.x;
    const bf16_t* p = W + (size_t)row * cols;
    float s = 0.0f;
    for (int c = tid; c < cols; c += 256) s += (float)p[c];
    red[tid] = s; __syncthreads();
    for (int off = 128; off; off >>= 1) {
        if (tid < off) red[tid] += red[tid + off];
        __syncthreads();
    }
    if (tid == 0) { const float d = 1.0f + red[0]; deg[row] = d; rinv[row] = rsqrtf(d); }
}

__global__ void k_fill(float* __restrict__ p, float v, int n)
{ const int i = blockIdx.x * blockDim.x + threadIdx.x; if (i < n) p[i] = v; }

// fp32 -> bf16, 4 elements/thread (float4 load, 8B packed store)
__global__ void k_cvt4(const float* __restrict__ in, bf16_t* __restrict__ out, int n4)
{
    const int i = blockIdx.x * blockDim.x + threadIdx.x;
    if (i >= n4) return;
    const float4 f = ((const float4*)in)[i];
    v4bf o; o[0] = (bf16_t)f.x; o[1] = (bf16_t)f.y; o[2] = (bf16_t)f.z; o[3] = (bf16_t)f.w;
    ((v4bf*)out)[i] = o;
}

// out[r*ldo + c0 + c] = in[r*64 + c]   (column-slice copy into concat buffer)
__global__ void k_cvt_cols(const float* __restrict__ in, bf16_t* __restrict__ out,
                           int ldo, int c0, int n)
{
    const int i = blockIdx.x * blockDim.x + threadIdx.x;
    if (i >= n) return;
    const int r = i >> 6, c = i & 63;
    out[(size_t)r * ldo + c0 + c] = (bf16_t)in[i];
}

// bf16 transpose: out[c*R + r] = in[r*C + c] * (rowscale ? rowscale[r] : 1)
__global__ __launch_bounds__(256)
void k_t_bf16(const float* __restrict__ in, bf16_t* __restrict__ out,
              int R, int C, const float* __restrict__ rowscale)
{
    __shared__ float tile[32][33];
    const int tx = threadIdx.x, ty = threadIdx.y;      // 32 x 8
    const int r0 = blockIdx.y * 32, c0 = blockIdx.x * 32;
#pragma unroll
    for (int i = 0; i < 32; i += 8) {
        const int r = r0 + ty + i, c = c0 + tx;
        float v = 0.0f;
        if (r < R && c < C) {
            v = in[(size_t)r * C + c];
            if (rowscale) v *= rowscale[r];
        }
        tile[ty + i][tx] = v;
    }
    __syncthreads();
#pragma unroll
    for (int i = 0; i < 32; i += 8) {
        const int c = c0 + ty + i, r = r0 + tx;
        if (c < C && r < R) out[(size_t)c * R + r] = (bf16_t)tile[tx][ty + i];
    }
}

// btOut[c*KEIG + k] = exp(-evals[k]*max(t[c],1e-8)) * spec[k*64 + c]
__global__ void k_diffuse_T(const float* __restrict__ spec, const float* __restrict__ evals,
                            const float* __restrict__ t, bf16_t* __restrict__ btOut)
{
    const int i = blockIdx.x * blockDim.x + threadIdx.x;
    if (i >= KEIG * DW) return;
    const int k = i & (KEIG - 1), c = i >> 7;
    const float tc = fmaxf(t[c], 1e-8f);
    btOut[(size_t)c * KEIG + k] = (bf16_t)(__expf(-evals[k] * tc) * spec[(size_t)k * 64 + c]);
}

// h[:,128:192] = tanh(gX*br + gY*bi), written straight into the bf16 concat.
__global__ void k_gradfeat(const float* __restrict__ gX, const float* __restrict__ gY,
                           const float* __restrict__ br, const float* __restrict__ bi,
                           bf16_t* __restrict__ h, int n)
{
    const int i = blockIdx.x * blockDim.x + threadIdx.x;
    if (i >= n) return;
    const int r = i >> 6, c = i & 63;
    const float f = tanhf(gX[i] * br[i] + gY[i] * bi[i]);
    h[(size_t)r * 192 + 128 + c] = (bf16_t)f;
}

__global__ void k_scatter_deg(const int* __restrict__ dst, const float* __restrict__ ew,
                              float* __restrict__ deg, int e)
{
    const int i = blockIdx.x * blockDim.x + threadIdx.x;
    if (i < e) unsafeAtomicAdd(&deg[dst[i]], ew[i]);
}

__global__ void k_dinv(const float* __restrict__ deg, float* __restrict__ dinv, int n)
{ const int i = blockIdx.x * blockDim.x + threadIdx.x; if (i < n) dinv[i] = rsqrtf(deg[i]); }

__global__ void k_edge_norm(const int* __restrict__ src, const int* __restrict__ dst,
                            const float* __restrict__ ew, const float* __restrict__ dinv,
                            float* __restrict__ norm, int e)
{
    const int i = blockIdx.x * blockDim.x + threadIdx.x;
    if (i < e) norm[i] = dinv[src[i]] * ew[i] * dinv[dst[i]];
}

// One block per edge, 64 lanes = channels: agg[dst] += norm * xw[src]
__global__ __launch_bounds__(64)
void k_gcn_agg(const int* __restrict__ src, const int* __restrict__ dst,
               const float* __restrict__ norm, const float* __restrict__ xw,
               float* __restrict__ agg)
{
    const int e = blockIdx.x, c = threadIdx.x;
    const int s = src[e], d = dst[e];
    unsafeAtomicAdd(&agg[(size_t)d * 64 + c], norm[e] * xw[(size_t)s * 64 + c]);
}

__global__ void k_gcn_fin(const float* __restrict__ agg, const float* __restrict__ xw,
                          const float* __restrict__ deg, const float* __restrict__ bias,
                          float* __restrict__ out, int relu, int n)
{
    const int i = blockIdx.x * blockDim.x + threadIdx.x;
    if (i >= n) return;
    const int r = i >> 6, c = i & 63;
    float v = agg[i] + xw[i] / deg[r] + bias[c];
    if (relu) v = fmaxf(v, 0.0f);
    out[i] = v;
}

// diff_x = 0.5*( xw_gs/deg_v + P*rsv + gs_b  +  xw_sg + sg_b )   (vertex rows)
__global__ void k_combine_v(const float* __restrict__ xwgs, const float* __restrict__ P,
                            const float* __restrict__ xwsg, const float* __restrict__ degv,
                            const float* __restrict__ rsv, const float* __restrict__ gsb,
                            const float* __restrict__ sgb, float* __restrict__ out, int n)
{
    const int i = blockIdx.x * blockDim.x + threadIdx.x;
    if (i >= n) return;
    const int r = i >> 6, c = i & 63;
    out[i] = 0.5f * (xwgs[i] / degv[r] + P[i] * rsv[r] + gsb[c] + xwsg[i] + sgb[c]);
}

// gx = 0.5*( xw_gs + gs_b  +  xw_sg/deg_g + Q*rsg + sg_b )       (graph rows)
__global__ void k_combine_g(const float* __restrict__ xwgs, const float* __restrict__ Q,
                            const float* __restrict__ xwsg, const float* __restrict__ degg,
                            const float* __restrict__ rsg, const float* __restrict__ gsb,
                            const float* __restrict__ sgb, float* __restrict__ out, int n)
{
    const int i = blockIdx.x * blockDim.x + threadIdx.x;
    if (i >= n) return;
    const int r = i >> 6, c = i & 63;
    out[i] = 0.5f * (xwgs[i] + gsb[c] + xwsg[i] / degg[r] + Q[i] * rsg[r] + sgb[c]);
}

// ------------------------------ host helpers --------------------------------

static inline void gemm64(hipStream_t s, const bf16_t* A, const bf16_t* BT, float* C,
                          const float* Cin, const float* bias, int M, int K, int S,
                          float alpha, int relu, int atomicOut)
{
    const int waves  = M / 16;
    const int blocks = (waves + 7) / 8;            // 8 waves / 256-thread block
    const int kChunk = K / S;                      // S chosen so 64 | kChunk
    k_gemm_n64<<<dim3(blocks, S), 256, 0, s>>>(A, BT, C, Cin, bias, M, K, kChunk,
                                               alpha, relu, atomicOut);
}

static inline void cvtT(hipStream_t s, const float* in, bf16_t* out, int R, int C,
                        const float* rowscale)
{
    dim3 g((C + 31) / 32, (R + 31) / 32), b(32, 8);
    k_t_bf16<<<g, b, 0, s>>>(in, out, R, C, rowscale);
}

static inline void cvt(hipStream_t s, const float* in, bf16_t* out, int n)
{ const int n4 = n / 4; k_cvt4<<<(n4 + 255) / 256, 256, 0, s>>>(in, out, n4); }

static inline void fill(hipStream_t s, float* p, float v, int n)
{ k_fill<<<(n + 255) / 256, 256, 0, s>>>(p, v, n); }

// ------------------------------- entry point --------------------------------

extern "C" void kernel_launch(void* const* d_in, const int* in_sizes, int n_in,
                              void* d_out, int out_size, void* d_ws, size_t ws_size,
                              hipStream_t stream)
{
    const float* vertices = (const float*)d_in[0];
    const float* graph_pos= (const float*)d_in[1];
    const float* x_in     = (const float*)d_in[2];
    const float* graph_x  = (const float*)d_in[3];
    const float* mass     = (const float*)d_in[4];
    const float* evals    = (const float*)d_in[5];
    const float* evecs    = (const float*)d_in[6];
    const float* gradX    = (const float*)d_in[7];
    const float* gradY    = (const float*)d_in[8];
    const int*   eidx     = (const int*)  d_in[9];
    const float* ew       = (const float*)d_in[10];
    const float* lin1_w   = (const float*)d_in[11];
    const float* lin1_b   = (const float*)d_in[12];
    const float* lin2_w   = (const float*)d_in[13];
    const float* lin2_b   = (const float*)d_in[14];
    const float* dtim     = (const float*)d_in[15];
    const float* A_re     = (const float*)d_in[16];
    const float* A_im     = (const float*)d_in[17];
    const float* w0 = (const float*)d_in[18]; const float* b0 = (const float*)d_in[19];
    const float* w1 = (const float*)d_in[20]; const float* b1 = (const float*)d_in[21];
    const float* w2 = (const float*)d_in[22]; const float* b2 = (const float*)d_in[23];
    const float* g1w= (const float*)d_in[24]; const float* g1b= (const float*)d_in[25];
    const float* g2w= (const float*)d_in[26]; const float* g2b= (const float*)d_in[27];
    const float* gsw= (const float*)d_in[28]; const float* gsb= (const float*)d_in[29];
    const float* sgw= (const float*)d_in[30]; const float* sgb= (const float*)d_in[31];
    const int* e_src = eidx;
    const int* e_dst = eidx + NE;

    // ---- workspace carve (bump allocator, 256B aligned) ----
    char* p = (char*)d_ws;
    auto carve = [&](size_t bytes) -> char* {
        char* r = p; p += (bytes + 255) & ~(size_t)255; return r;
    };
    bf16_t* Wb_b    = (bf16_t*)carve((size_t)NV * NG * 2);
    bf16_t* WbT_b   = (bf16_t*)carve((size_t)NV * NG * 2);
    bf16_t* gradX_b = (bf16_t*)carve((size_t)NV * NV * 2);
    bf16_t* gradY_b = (bf16_t*)carve((size_t)NV * NV * 2);
    bf16_t* evecs_b = (bf16_t*)carve((size_t)NV * KEIG * 2);
    bf16_t* evecsT_b= (bf16_t*)carve((size_t)NV * KEIG * 2);
    bf16_t* xin_b   = (bf16_t*)carve((size_t)NV * CIN * 2);
    bf16_t* gxin_b  = (bf16_t*)carve((size_t)NG * CIN * 2);
    float*  diff_x  = (float*) carve((size_t)NV * 64 * 4);
    float*  gxbuf   = (float*) carve((size_t)NG * 64 * 4);
    float*  deg_v   = (float*) carve((size_t)NV * 4);
    float*  rsv     = (float*) carve((size_t)NV * 4);
    float*  deg_g   = (float*) carve((size_t)NG * 4);
    float*  rsg     = (float*) carve((size_t)NG * 4);
    float*  deg_e   = (float*) carve((size_t)NG * 4);
    float*  dinv_e  = (float*) carve((size_t)NG * 4);
    float*  enorm   = (float*) carve((size_t)NE * 4);
    bf16_t* bt_mass = (bf16_t*)carve((size_t)64 * NV * 2);
    bf16_t* bt_spec = (bf16_t*)carve((size_t)64 * KEIG * 2);
    bf16_t* bt_xd   = (bf16_t*)carve((size_t)64 * NV * 2);
    bf16_t* gX_b    = (bf16_t*)carve((size_t)NV * 64 * 2);
    bf16_t* gY_b    = (bf16_t*)carve((size_t)NV * 64 * 2);
    float*  spec    = (float*) carve((size_t)KEIG * 64 * 4);
    float*  xd      = (float*) carve((size_t)NV * 64 * 4);
    float*  gXf     = (float*) carve((size_t)NV * 64 * 4);
    float*  gYf     = (float*) carve((size_t)NV * 64 * 4);
    float*  br      = (float*) carve((size_t)NV * 64 * 4);
    float*  bi      = (float*) carve((size_t)NV * 64 * 4);
    bf16_t* h_b     = (bf16_t*)carve((size_t)NV * 192 * 2);
    float*  h1      = (float*) carve((size_t)NV * 64 * 4);
    bf16_t* h1_b    = (bf16_t*)carve((size_t)NV * 64 * 2);
    float*  h2      = (float*) carve((size_t)NV * 64 * 4);
    bf16_t* h2_b    = (bf16_t*)carve((size_t)NV * 64 * 2);
    float*  xw      = (float*) carve((size_t)NG * 64 * 4);
    float*  agg     = (float*) carve((size_t)NG * 64 * 4);
    float*  gx1     = (float*) carve((size_t)NG * 64 * 4);
    bf16_t* gx_b    = (bf16_t*)carve((size_t)NG * 64 * 2);
    bf16_t* gx1_b   = (bf16_t*)carve((size_t)NG * 64 * 2);
    bf16_t* xall_b  = (bf16_t*)carve((size_t)(NV + NG) * 64 * 2);
    float*  xw_gs   = (float*) carve((size_t)(NV + NG) * 64 * 4);
    float*  xw_sg   = (float*) carve((size_t)(NV + NG) * 64 * 4);
    bf16_t* bt_g    = (bf16_t*)carve((size_t)64 * NG * 2);
    bf16_t* bt_v    = (bf16_t*)carve((size_t)64 * NV * 2);
    float*  Pm      = (float*) carve((size_t)NV * 64 * 4);
    float*  Qm      = (float*) carve((size_t)NG * 64 * 4);
    bf16_t* wt0 = (bf16_t*)carve(64 * 192 * 2);
    bf16_t* wt1 = (bf16_t*)carve(64 * 192 * 2);
    bf16_t* wt2 = (bf16_t*)carve(64 * 192 * 2);
    bf16_t* wt3 = (bf16_t*)carve(64 * 192 * 2);
    bf16_t* wt4 = (bf16_t*)carve(64 * 192 * 2);

    // ---- one-time prep ----
    { dim3 g(NG / 32, NV / 32), b(32, 8);
      k_wb<<<g, b, 0, stream>>>(vertices, graph_pos, Wb_b, WbT_b); }
    k_rowsum<<<NV, 256, 0, stream>>>(Wb_b,  NG, deg_v, rsv);
    k_rowsum<<<NG, 256, 0, stream>>>(WbT_b, NV, deg_g, rsg);
    cvt(stream, gradX, gradX_b, NV * NV);
    cvt(stream, gradY, gradY_b, NV * NV);
    cvt(stream, evecs, evecs_b, NV * KEIG);
    cvtT(stream, evecs, evecsT_b, NV, KEIG, nullptr);       // [KEIG, NV]
    cvt(stream, x_in, xin_b, NV * CIN);
    cvt(stream, graph_x, gxin_b, NG * CIN);

    // edge-degree / norm (block-invariant)
    fill(stream, deg_e, 1.0f, NG);
    k_scatter_deg<<<(NE + 255) / 256, 256, 0, stream>>>(e_dst, ew, deg_e, NE);
    k_dinv<<<(NG + 255) / 256, 256, 0, stream>>>(deg_e, dinv_e, NG);
    k_edge_norm<<<(NE + 255) / 256, 256, 0, stream>>>(e_src, e_dst, ew, dinv_e, enorm, NE);

    // diff_x = x_in @ lin1_w + b ;  gx = graph_x @ lin2_w + b
    cvtT(stream, lin1_w, wt0, CIN, 64, nullptr);
    gemm64(stream, xin_b, wt0, diff_x, nullptr, lin1_b, NV, CIN, 1, 1.0f, 0, 0);
    cvtT(stream, lin2_w, wt1, CIN, 64, nullptr);
    gemm64(stream, gxin_b, wt1, gxbuf, nullptr, lin2_b, NG, CIN, 1, 1.0f, 0, 0);

    const int nV64 = NV * 64, nG64 = NG * 64;

    for (int i = 0; i < NBLK; ++i) {
        // ---- DiffusionNet spectral step ----
        cvtT(stream, diff_x, bt_mass, NV, 64, mass);                 // (x*m)^T
        fill(stream, spec, 0.0f, KEIG * 64);
        gemm64(stream, evecsT_b, bt_mass, spec, nullptr, nullptr, KEIG, NV,
               64, 1.0f, 0, 1);                                      // split-K x64
        k_diffuse_T<<<(KEIG * 64 + 255) / 256, 256, 0, stream>>>(spec, evals,
                                                                 dtim + i * 64, bt_spec);
        gemm64(stream, evecs_b, bt_spec, xd, nullptr, nullptr, NV, KEIG, 1, 1.0f, 0, 0);
        cvtT(stream, xd, bt_xd, NV, 64, nullptr);
        fill(stream, gXf, 0.0f, nV64);
        fill(stream, gYf, 0.0f, nV64);
        gemm64(stream, gradX_b, bt_xd, gXf, nullptr, nullptr, NV, NV, 4, 1.0f, 0, 1);
        gemm64(stream, gradY_b, bt_xd, gYf, nullptr, nullptr, NV, NV, 4, 1.0f, 0, 1);
        cvt(stream, gXf, gX_b, nV64);
        cvt(stream, gYf, gY_b, nV64);
        cvtT(stream, A_re + (size_t)i * 64 * 64, wt0, 64, 64, nullptr);
        cvtT(stream, A_im + (size_t)i * 64 * 64, wt1, 64, 64, nullptr);
        gemm64(stream, gX_b, wt0, br, nullptr, nullptr, NV, 64, 1,  1.0f, 0, 0);
        gemm64(stream, gY_b, wt1, br, br,      nullptr, NV, 64, 1, -1.0f, 0, 0);
        gemm64(stream, gY_b, wt0, bi, nullptr, nullptr, NV, 64, 1,  1.0f, 0, 0);
        gemm64(stream, gX_b, wt1, bi, bi,      nullptr, NV, 64, 1,  1.0f, 0, 0);
        // h = [diff_x | xd | tanh(gX*br + gY*bi)]  (bf16, ld=192)
        k_cvt_cols<<<(nV64 + 255) / 256, 256, 0, stream>>>(diff_x, h_b, 192, 0,  nV64);
        k_cvt_cols<<<(nV64 + 255) / 256, 256, 0, stream>>>(xd,     h_b, 192, 64, nV64);
        k_gradfeat<<<(nV64 + 255) / 256, 256, 0, stream>>>(gXf, gYf, br, bi, h_b, nV64);
        cvtT(stream, w0 + (size_t)i * 192 * 64, wt2, 192, 64, nullptr);
        gemm64(stream, h_b, wt2, h1, nullptr, b0 + i * 64, NV, 192, 1, 1.0f, 1, 0);
        cvt(stream, h1, h1_b, nV64);
        cvtT(stream, w1 + (size_t)i * 64 * 64, wt3, 64, 64, nullptr);
        gemm64(stream, h1_b, wt3, h2, nullptr, b1 + i * 64, NV, 64, 1, 1.0f, 1, 0);
        cvt(stream, h2, h2_b, nV64);
        cvtT(stream, w2 + (size_t)i * 64 * 64, wt4, 64, 64, nullptr);
        gemm64(stream, h2_b, wt4, diff_x, diff_x, b2 + i * 64, NV, 64, 1, 1.0f, 0, 0);

        // ---- internal graph GCN x2 ----
        cvt(stream, gxbuf, gx_b, nG64);
        cvtT(stream, g1w + (size_t)i * 64 * 64, wt0, 64, 64, nullptr);
        gemm64(stream, gx_b, wt0, xw, nullptr, nullptr, NG, 64, 1, 1.0f, 0, 0);
        fill(stream, agg, 0.0f, nG64);
        k_gcn_agg<<<NE, 64, 0, stream>>>(e_src, e_dst, enorm, xw, agg);
        k_gcn_fin<<<(nG64 + 255) / 256, 256, 0, stream>>>(agg, xw, deg_e,
                                                          g1b + i * 64, gx1, 1, nG64);
        cvt(stream, gx1, gx1_b, nG64);
        cvtT(stream, g2w + (size_t)i * 64 * 64, wt1, 64, 64, nullptr);
        gemm64(stream, gx1_b, wt1, xw, nullptr, nullptr, NG, 64, 1, 1.0f, 0, 0);
        fill(stream, agg, 0.0f, nG64);
        k_gcn_agg<<<NE, 64, 0, stream>>>(e_src, e_dst, enorm, xw, agg);
        k_gcn_fin<<<(nG64 + 255) / 256, 256, 0, stream>>>(agg, xw, deg_e,
                                                          g2b + i * 64, gxbuf, 0, nG64);

        // ---- bipartite GCNs ----
        cvt(stream, diff_x, xall_b, nV64);
        cvt(stream, gxbuf, xall_b + (size_t)NV * 64, nG64);
        cvtT(stream, gsw + (size_t)i * 64 * 64, wt2, 64, 64, nullptr);
        cvtT(stream, sgw + (size_t)i * 64 * 64, wt3, 64, 64, nullptr);
        gemm64(stream, xall_b, wt2, xw_gs, nullptr, nullptr, NV + NG, 64, 1, 1.0f, 0, 0);
        gemm64(stream, xall_b, wt3, xw_sg, nullptr, nullptr, NV + NG, 64, 1, 1.0f, 0, 0);
        cvtT(stream, xw_gs + (size_t)NV * 64, bt_g, NG, 64, nullptr);
        fill(stream, Pm, 0.0f, nV64);
        gemm64(stream, Wb_b, bt_g, Pm, nullptr, nullptr, NV, NG, 4, 1.0f, 0, 1);
        cvtT(stream, xw_sg, bt_v, NV, 64, nullptr);
        fill(stream, Qm, 0.0f, nG64);
        gemm64(stream, WbT_b, bt_v, Qm, nullptr, nullptr, NG, NV, 8, 1.0f, 0, 1);
        k_combine_v<<<(nV64 + 255) / 256, 256, 0, stream>>>(xw_gs, Pm, xw_sg, deg_v, rsv,
                                                            gsb + i * 64, sgb + i * 64,
                                                            diff_x, nV64);
        k_combine_g<<<(nG64 + 255) / 256, 256, 0, stream>>>(xw_gs + (size_t)NV * 64, Qm,
                                                            xw_sg + (size_t)NV * 64, deg_g,
                                                            rsg, gsb + i * 64, sgb + i * 64,
                                                            gxbuf, nG64);
    }

    hipMemcpyAsync(d_out, diff_x, (size_t)out_size * sizeof(float),
                   hipMemcpyDeviceToDevice, stream);
}
